// LiquidEmber_16655883174426
// MI455X (gfx1250) — matrix operand order repaired and checked by reading, not compiled, not run on hardware
//
#include <hip/hip_runtime.h>
#include <hip/hip_bf16.h>
#include <math.h>

// ---------------------------------------------------------------------------
// Model dims (fixed by the reference)
// ---------------------------------------------------------------------------
constexpr int Bsz = 8, Tsz = 4096, Dsz = 512, Usz = 768, Lsz = 4, Vsz = 256, Fsz = 2048;
constexpr int NTOK = Bsz * Tsz;           // 32768 rows in the token dimension
constexpr int NPK  = 4 * Usz;             // 3072 packed projection columns
constexpr int WROW = Dsz + Usz;           // 1280 rows of each W* matrix

typedef _Float16 half_t;
typedef __attribute__((ext_vector_type(16))) _Float16 v16h;
typedef __attribute__((ext_vector_type(8)))  _Float16 v8h;
typedef __attribute__((ext_vector_type(8)))  float    v8f;

union V16 { v16h v; v8h h[2]; };

__device__ __forceinline__ v8f wmma32(v16h a, v16h b, v8f c) {
    // D = A(16x32 f16) * B(32x16 f16) + C(16x16 f32)
    return __builtin_amdgcn_wmma_f32_16x16x32_f16(false, a, false, b, (short)0, c, false, false);
}
__device__ __forceinline__ float sigm(float x)  { return 1.0f / (1.0f + expf(-x)); }
__device__ __forceinline__ float gelu_exact(float x) {
    return 0.5f * x * (1.0f + erff(x * 0.70710678118654752f));
}

// ---------------------------------------------------------------------------
// CDNA5 async global->LDS copy (bypasses VGPRs, tracked by ASYNCcnt).
// LDS destination = low 32 bits of the flat shared-pointer (LDS aperture maps
// addr[31:0] to the wave's LDS offset). Global address in a VGPR pair.
// ---------------------------------------------------------------------------
__device__ __forceinline__ void async_copy_b128(void* lds_dst, const void* gsrc) {
    unsigned int       ldsoff = (unsigned int)(uintptr_t)lds_dst;
    unsigned long long gaddr  = (unsigned long long)(uintptr_t)gsrc;
    asm volatile("global_load_async_to_lds_b128 %0, %1, off"
                 :: "v"(ldsoff), "v"(gaddr) : "memory");
}
__device__ __forceinline__ void wait_async0() {
#if __has_builtin(__builtin_amdgcn_s_wait_asynccnt)
    __builtin_amdgcn_s_wait_asynccnt(0);
#else
    asm volatile("s_wait_asynccnt 0" ::: "memory");
#endif
}

// ---------------------------------------------------------------------------
// Generic WMMA GEMM: C[M,N] = A[M,K] @ B[K,N] (+bias, epilogue variants)
// A: f16 row-major (lda=K). B: f16 row-major (ldb=N), or (N,K) row-major if BT.
// Block = 256 threads = 8 waves; macro tile 64(M) x 128(N); K-step 32.
// Each wave computes a 32x32 sub-tile = 4 WMMA accumulators.
// Double-buffered LDS tiles; A (and BT-form B) staged with async-to-LDS copies
// overlapped with the WMMAs of the current buffer. All dims are multiples of
// the tile sizes -> no bounds predication (WMMA requires EXEC all-ones anyway).
// ---------------------------------------------------------------------------
enum { EPI_F32 = 0, EPI_GELU_F16 = 1, EPI_RESID_F32 = 2, EPI_F16 = 3 };

template<int EPI, int BT>
__global__ __launch_bounds__(256)
void gemm_wmma_f16(const half_t* __restrict__ A, const half_t* __restrict__ B,
                   const float* __restrict__ bias, const float* __restrict__ resid,
                   float* __restrict__ Cf, half_t* __restrict__ Ch,
                   int M, int N, int K)
{
    // Row stride 40 halves (80B): de-conflicts b128 LDS reads. Two buffers.
    __shared__ __align__(16) half_t As[2][64][40];
    __shared__ __align__(16) half_t Bs[2][128][40];   // stored [n][k] (transposed)

    const int tid  = threadIdx.x;
    const int lane = tid & 31;
    const int wid  = tid >> 5;
    const int m0   = blockIdx.y * 64;
    const int n0   = blockIdx.x * 128;
    const int wm   = (wid >> 2) * 32;              // wave M offset in macro tile
    const int wn   = (wid & 3) * 32;               // wave N offset
    const int hi   = lane >> 4;                    // lane half-group (K select)
    const int lr   = lane & 15;                    // row/col within fragment

    auto stageA = [&](int buf, int k0) {
        int row = tid >> 2, kc = (tid & 3) * 8;
        async_copy_b128(&As[buf][row][kc], A + (size_t)(m0 + row) * K + k0 + kc);
    };
    auto stageB = [&](int buf, int k0) {
        if (BT) {
            #pragma unroll
            for (int p = 0; p < 2; ++p) {
                int n = (tid >> 2) + p * 64, kc = (tid & 3) * 8;
                async_copy_b128(&Bs[buf][n][kc], B + (size_t)(n0 + n) * K + k0 + kc);
            }
        } else {
            #pragma unroll
            for (int p = 0; p < 2; ++p) {
                int r = (tid >> 4) + p * 16, c = (tid & 15) * 8;
                const half_t* bp = B + (size_t)(k0 + r) * N + n0 + c;
                if (k0 + 32 < K) __builtin_prefetch(bp + (size_t)32 * N, 0, 1);
                v8h bv = *(const v8h*)bp;
                #pragma unroll
                for (int j = 0; j < 8; ++j) Bs[buf][c + j][r] = bv[j];
            }
        }
    };

    v8f acc[2][2];
    #pragma unroll
    for (int i = 0; i < 2; ++i)
        #pragma unroll
        for (int j = 0; j < 2; ++j) acc[i][j] = (v8f){};

    // prologue: stage first K-tile
    stageA(0, 0);
    stageB(0, 0);
    wait_async0();
    __syncthreads();

    int buf = 0;
    for (int k0 = 0; k0 < K; k0 += 32) {
        // kick off async staging of the next K-tile into the other buffer
        if (k0 + 32 < K) { stageA(buf ^ 1, k0 + 32); stageB(buf ^ 1, k0 + 32); }

        // ---- fragments + 4 WMMAs per wave on current buffer ----
        V16 af[2], bf[2];
        #pragma unroll
        for (int mi = 0; mi < 2; ++mi) {
            int m = wm + mi * 16 + lr;
            af[mi].h[0] = *(const v8h*)&As[buf][m][8 * hi];        // K = 8*hi + 0..7
            af[mi].h[1] = *(const v8h*)&As[buf][m][16 + 8 * hi];   // K = 16+8*hi + 0..7
        }
        #pragma unroll
        for (int ni = 0; ni < 2; ++ni) {
            int n = wn + ni * 16 + lr;
            bf[ni].h[0] = *(const v8h*)&Bs[buf][n][8 * hi];        // K = 8*hi + 0..7
            bf[ni].h[1] = *(const v8h*)&Bs[buf][n][16 + 8 * hi];   // K = 16+8*hi + 0..7
        }
        #pragma unroll
        for (int mi = 0; mi < 2; ++mi)
            #pragma unroll
            for (int ni = 0; ni < 2; ++ni)
                acc[mi][ni] = wmma32(af[mi].v, bf[ni].v, acc[mi][ni]);

        wait_async0();          // next tile landed in LDS
        __syncthreads();        // visible to all waves; current tile fully read
        buf ^= 1;
    }

    // ---- epilogue: C layout VGPR r, lanes<16 -> M=r, lanes>=16 -> M=r+8 ----
    #pragma unroll
    for (int mi = 0; mi < 2; ++mi) {
        #pragma unroll
        for (int ni = 0; ni < 2; ++ni) {
            int mbase = m0 + wm + mi * 16 + 8 * hi;
            int n     = n0 + wn + ni * 16 + lr;
            float bv  = bias ? bias[n] : 0.0f;
            #pragma unroll
            for (int r = 0; r < 8; ++r) {
                size_t o = (size_t)(mbase + r) * N + n;
                float v = acc[mi][ni][r] + bv;
                if      (EPI == EPI_F32)      Cf[o] = v;
                else if (EPI == EPI_GELU_F16) Ch[o] = (half_t)gelu_exact(v);
                else if (EPI == EPI_RESID_F32)Cf[o] = resid[o] + v;
                else                          Ch[o] = (half_t)v;
            }
        }
    }
}

// ---------------------------------------------------------------------------
// CfC recurrent scan. Persistent kernel: grid = 16 WGs x 96 threads (3 waves).
// WG g owns U-columns [48g, 48g+48); its 768x192 gate-interleaved weight slab
// (294 KB f16) lives in LDS for the whole scan (CDNA5 320KB LDS/WGP).
// Per step: h(16x768, rows 8..15 zero) @ slab via WMMA; wave w owns one
// matched (ff1,ff2,ta,tb) tile group of 16 U-cols -> gating in registers.
// Double-buffered h in global (L2-resident); global atomic barrier per step.
// ---------------------------------------------------------------------------
__global__ __launch_bounds__(96)
void cfc_scan(const half_t* __restrict__ Wrec,   // rows 512..1279 of packed Wcat[i]
              const half_t* __restrict__ proj,   // (NTOK, 3072) f16, gate-interleaved
              half_t* __restrict__ hbuf,         // [2][16][768] f16, pre-zeroed
              float* __restrict__ cfc,           // (NTOK, 512) f32 out
              unsigned int* __restrict__ ctr)    // pre-zeroed barrier counter
{
    extern __shared__ __align__(16) half_t Bsl[];  // [192][776]
    constexpr int LDB = 776;                       // padded K stride (halves)
    const int g    = blockIdx.x;                   // 0..15: U-slice
    const int tid  = threadIdx.x;
    const int lane = tid & 31;
    const int wid  = tid >> 5;                     // 0..2
    const int hi   = lane >> 4, lr = lane & 15;

    // Load weight slab once, transposed to Bsl[n_local][k].
    for (int it = tid; it < 768 * 24; it += 96) {
        int k = it / 24, c8 = (it % 24) * 8;
        v8h w = *(const v8h*)(Wrec + (size_t)k * NPK + g * 192 + c8);
        #pragma unroll
        for (int j = 0; j < 8; ++j) Bsl[(size_t)(c8 + j) * LDB + k] = w[j];
    }
    __syncthreads();

    for (int t = 0; t < Tsz; ++t) {
        const half_t* hin  = hbuf + (size_t)(t & 1) * 16 * 768;
        half_t*       hout = hbuf + (size_t)((t + 1) & 1) * 16 * 768;

        v8f acc[4];
        #pragma unroll
        for (int q = 0; q < 4; ++q) acc[q] = (v8f){};

        for (int k0 = 0; k0 < Usz; k0 += 32) {
            V16 a;
            const half_t* hp = hin + (size_t)lr * 768 + k0;
            a.h[0] = *(const v8h*)(hp + 8 * hi);
            a.h[1] = *(const v8h*)(hp + 16 + 8 * hi);
            #pragma unroll
            for (int q = 0; q < 4; ++q) {          // ff1, ff2, ta, tb tiles
                int n = wid * 64 + q * 16 + lr;
                V16 b;
                b.h[0] = *(const v8h*)&Bsl[(size_t)n * LDB + k0 + 8 * hi];
                b.h[1] = *(const v8h*)&Bsl[(size_t)n * LDB + k0 + 16 + 8 * hi];
                acc[q] = wmma32(a.v, b.v, acc[q]);
            }
        }

        if (lane < 16) {                            // rows 0..7 = batch rows
            int npack = g * 192 + wid * 64;         // packed col base (gate*16+lr)
            int j     = g * 48 + wid * 16 + lr;     // true U column
            #pragma unroll
            for (int r = 0; r < 8; ++r) {
                size_t prow = ((size_t)r * Tsz + t) * NPK + npack;
                float x1 = (float)proj[prow + 0 * 16 + lr];
                float x2 = (float)proj[prow + 1 * 16 + lr];
                float xa = (float)proj[prow + 2 * 16 + lr];
                float xb = (float)proj[prow + 3 * 16 + lr];
                float ff1 = tanhf(acc[0][r] + x1);
                float ff2 = tanhf(acc[1][r] + x2);
                float ti  = sigm((acc[2][r] + xa) + (acc[3][r] + xb));
                float hn  = ff1 * (1.0f - ti) + ti * ff2;
                hout[(size_t)r * 768 + j] = (half_t)hn;
                if (j < Dsz) cfc[((size_t)r * Tsz + t) * Dsz + j] = hn;
            }
        }

        // grid-wide step barrier (16 arrivals, monotone counter)
        __syncthreads();
        if (tid == 0) {
            __threadfence();
            atomicAdd(ctr, 1u);
            unsigned int target = 16u * (unsigned int)(t + 1);
            while (atomicAdd(ctr, 0u) < target) { __builtin_amdgcn_s_sleep(1); }
            __threadfence();
        }
        __syncthreads();
    }
}

// ---------------------------------------------------------------------------
// Small kernels
// ---------------------------------------------------------------------------
__global__ void embed_gather(const int* __restrict__ idx, const float* __restrict__ wte,
                             float* __restrict__ x)
{
    int bt = blockIdx.x;
    ((float4*)(x + (size_t)bt * Dsz))[threadIdx.x] =
        ((const float4*)(wte + (size_t)idx[bt] * Dsz))[threadIdx.x];
}

__global__ __launch_bounds__(256)
void layernorm_f16(const float* __restrict__ x, const float* __restrict__ s,
                   const float* __restrict__ b, half_t* __restrict__ out)
{
    int lane = threadIdx.x & 31, wid = threadIdx.x >> 5;
    size_t row = (size_t)blockIdx.x * 8 + wid;
    const float* xr = x + row * Dsz;
    float v[16], sum = 0.f, sq = 0.f;
    #pragma unroll
    for (int i = 0; i < 16; ++i) { v[i] = xr[lane + i * 32]; sum += v[i]; sq += v[i] * v[i]; }
    #pragma unroll
    for (int o = 16; o; o >>= 1) { sum += __shfl_xor(sum, o, 32); sq += __shfl_xor(sq, o, 32); }
    float mean = sum * (1.0f / Dsz);
    float inv  = rsqrtf(sq * (1.0f / Dsz) - mean * mean + 1e-5f);
    #pragma unroll
    for (int i = 0; i < 16; ++i) {
        int d = lane + i * 32;
        out[row * Dsz + d] = (half_t)((v[i] - mean) * inv * s[d] + b[d]);
    }
}

__global__ void lif_residual(float* __restrict__ x, const float* __restrict__ cfc,
                             const float* __restrict__ thr_p, const float* __restrict__ leak_p,
                             const float* __restrict__ steep_p, size_t n)
{
    size_t i = (size_t)blockIdx.x * blockDim.x + threadIdx.x;
    if (i >= n) return;
    int d = (int)(i & (Dsz - 1));
    float c     = cfc[i];
    float thr   = fabsf(thr_p[d]) * 0.1f;
    float leak  = sigm(leak_p[d]);
    float steep = log1pf(expf(steep_p[d]));
    float fire  = sigm(steep * (fabsf(c) - thr));
    float gate  = fire + leak * (1.0f - fire);
    x[i] += c * gate;
}

// Gate-interleaved packing: packed col n = 64*(j/16) + 16*gate + (j%16)
__global__ void pack_wcat(const float* __restrict__ W1, const float* __restrict__ W2,
                          const float* __restrict__ Wa, const float* __restrict__ Wb,
                          half_t* __restrict__ dst)
{
    size_t i = (size_t)blockIdx.x * blockDim.x + threadIdx.x;
    if (i >= (size_t)Lsz * WROW * NPK) return;
    int n = (int)(i % NPK); size_t ki = i / NPK;
    int k = (int)(ki % WROW); int layer = (int)(ki / WROW);
    int grp = n >> 6, gate = (n >> 4) & 3, c = n & 15;
    int j = grp * 16 + c;
    const float* src = (gate == 0) ? W1 : (gate == 1) ? W2 : (gate == 2) ? Wa : Wb;
    dst[i] = (half_t)src[((size_t)layer * WROW + k) * Usz + j];
}

__global__ void pack_bcat(const float* __restrict__ b1, const float* __restrict__ b2,
                          const float* __restrict__ ba, const float* __restrict__ bb,
                          float* __restrict__ dst)
{
    size_t i = (size_t)blockIdx.x * blockDim.x + threadIdx.x;
    if (i >= (size_t)Lsz * NPK) return;
    int n = (int)(i % NPK); int layer = (int)(i / NPK);
    int grp = n >> 6, gate = (n >> 4) & 3, c = n & 15;
    int j = grp * 16 + c;
    const float* src = (gate == 0) ? b1 : (gate == 1) ? b2 : (gate == 2) ? ba : bb;
    dst[i] = src[(size_t)layer * Usz + j];
}

__global__ void cvt_f32_f16(const float* __restrict__ src, half_t* __restrict__ dst, size_t n)
{
    size_t i = (size_t)blockIdx.x * blockDim.x + threadIdx.x;
    if (i < n) dst[i] = (half_t)src[i];
}

__global__ void zero_u32(unsigned int* __restrict__ p, size_t n)
{
    size_t i = (size_t)blockIdx.x * blockDim.x + threadIdx.x;
    if (i < n) p[i] = 0u;
}

__global__ __launch_bounds__(256)
void nll_loss(const float* __restrict__ logits, const int* __restrict__ tgt,
              float* __restrict__ acc)
{
    int lane = threadIdx.x & 31, wid = threadIdx.x >> 5;
    size_t row = (size_t)blockIdx.x * 8 + wid;
    const float* lr = logits + row * Vsz;
    float v[8], m = -INFINITY;
    #pragma unroll
    for (int i = 0; i < 8; ++i) { v[i] = lr[lane + i * 32]; m = fmaxf(m, v[i]); }
    #pragma unroll
    for (int o = 16; o; o >>= 1) m = fmaxf(m, __shfl_xor(m, o, 32));
    float s = 0.f;
    #pragma unroll
    for (int i = 0; i < 8; ++i) s += expf(v[i] - m);
    #pragma unroll
    for (int o = 16; o; o >>= 1) s += __shfl_xor(s, o, 32);
    if (lane == 0) {
        int t = tgt[row];
        int valid = (t != -1);
        float lp = lr[valid ? t : 0] - m - logf(s);
        atomicAdd(&acc[0], valid ? -lp : 0.0f);
        atomicAdd(&acc[1], (float)valid);
    }
}

__global__ void loss_finalize(const float* __restrict__ acc, float* __restrict__ out)
{
    out[0] = acc[0] / fmaxf(acc[1], 1.0f);
}

// ---------------------------------------------------------------------------
// Host orchestration
// ---------------------------------------------------------------------------
extern "C" void kernel_launch(void* const* d_in, const int* in_sizes, int n_in,
                              void* d_out, int out_size, void* d_ws, size_t ws_size,
                              hipStream_t stream)
{
    const int*   idx     = (const int*)  d_in[0];
    const int*   targets = (const int*)  d_in[1];
    const float* wte     = (const float*)d_in[2];
    const float* ln1_s   = (const float*)d_in[3];
    const float* ln1_b   = (const float*)d_in[4];
    const float* Wff1    = (const float*)d_in[5];
    const float* bff1    = (const float*)d_in[6];
    const float* Wff2    = (const float*)d_in[7];
    const float* bff2    = (const float*)d_in[8];
    const float* Wta     = (const float*)d_in[9];
    const float* bta     = (const float*)d_in[10];
    const float* Wtb     = (const float*)d_in[11];
    const float* btb     = (const float*)d_in[12];
    const float* lif_thr = (const float*)d_in[13];
    const float* lif_lk  = (const float*)d_in[14];
    const float* lif_st  = (const float*)d_in[15];
    const float* ln2_s   = (const float*)d_in[16];
    const float* ln2_b   = (const float*)d_in[17];
    const float* mlp_W1  = (const float*)d_in[18];
    const float* mlp_b1  = (const float*)d_in[19];
    const float* mlp_W2  = (const float*)d_in[20];
    const float* mlp_b2  = (const float*)d_in[21];
    const float* lnf_s   = (const float*)d_in[22];
    const float* lnf_b   = (const float*)d_in[23];
    float* out = (float*)d_out;                   // [NTOK*V logits | loss]

    // -------- workspace carve-up --------
    char* ws = (char*)d_ws; size_t off = 0;
    auto carve = [&](size_t bytes) -> void* {
        off = (off + 255) & ~(size_t)255;
        void* p = ws + off; off += bytes; return p;
    };
    float*  x    = (float*) carve((size_t)NTOK * Dsz * 4);            // residual stream
    float*  cfc  = (float*) carve((size_t)NTOK * Dsz * 4);            // scan output
    half_t* nrm  = (half_t*)carve((size_t)NTOK * Dsz * 2);            // normed (f16)
    half_t* proj = (half_t*)carve((size_t)NTOK * NPK * 2);            // x1|x2|xa|xb packed
    half_t* act1 = (half_t*)carve((size_t)NTOK * Fsz * 2);            // gelu(mlp1)
    half_t* Wcat = (half_t*)carve((size_t)Lsz * WROW * NPK * 2);      // packed f16 weights
    float*  bcat = (float*) carve((size_t)Lsz * NPK * 4);
    half_t* W1h  = (half_t*)carve((size_t)Lsz * Dsz * Fsz * 2);
    half_t* W2h  = (half_t*)carve((size_t)Lsz * Fsz * Dsz * 2);
    half_t* wteh = (half_t*)carve((size_t)Vsz * Dsz * 2);
    half_t* hbuf = (half_t*)carve((size_t)2 * 16 * Usz * 2);          // double-buffered h
    unsigned int* ctr = (unsigned int*)carve(256);
    float* lacc = (float*)carve(256);
    (void)ws_size; (void)n_in; (void)in_sizes; (void)out_size;

    // -------- weight conversion / packing (deterministic each call) --------
    {
        size_t n = (size_t)Lsz * WROW * NPK;
        pack_wcat<<<(unsigned)((n + 255) / 256), 256, 0, stream>>>(Wff1, Wff2, Wta, Wtb, Wcat);
        size_t nb = (size_t)Lsz * NPK;
        pack_bcat<<<(unsigned)((nb + 255) / 256), 256, 0, stream>>>(bff1, bff2, bta, btb, bcat);
        size_t n1 = (size_t)Lsz * Dsz * Fsz;
        cvt_f32_f16<<<(unsigned)((n1 + 255) / 256), 256, 0, stream>>>(mlp_W1, W1h, n1);
        size_t n2 = (size_t)Lsz * Fsz * Dsz;
        cvt_f32_f16<<<(unsigned)((n2 + 255) / 256), 256, 0, stream>>>(mlp_W2, W2h, n2);
        size_t nw = (size_t)Vsz * Dsz;
        cvt_f32_f16<<<(unsigned)((nw + 255) / 256), 256, 0, stream>>>(wte, wteh, nw);
    }

    // -------- embedding --------
    embed_gather<<<NTOK, Dsz / 4, 0, stream>>>(idx, wte, x);

    const size_t scanLds = (size_t)192 * 776 * sizeof(half_t);   // 298 KB weight slab

    for (int i = 0; i < Lsz; ++i) {
        // ln1 -> f16
        layernorm_f16<<<NTOK / 8, 256, 0, stream>>>(x, ln1_s + (size_t)i * Dsz,
                                                    ln1_b + (size_t)i * Dsz, nrm);
        // input projections: (32768x512) @ (512x3072) packed, +bias -> f16
        gemm_wmma_f16<EPI_F16, 0><<<dim3(NPK / 128, NTOK / 64), 256, 0, stream>>>(
            nrm, Wcat + (size_t)i * WROW * NPK, bcat + (size_t)i * NPK,
            nullptr, nullptr, proj, NTOK, NPK, Dsz);
        // reset h double-buffer (h0 = 0, rows 8..15 stay 0) and step barrier
        zero_u32<<<((2 * 16 * Usz * 2 / 4) + 255) / 256, 256, 0, stream>>>(
            (unsigned int*)hbuf, 2 * 16 * Usz * 2 / 4);
        zero_u32<<<1, 32, 0, stream>>>(ctr, 1);
        // sequential CfC scan (persistent, LDS-resident weights)
        cfc_scan<<<16, 96, scanLds, stream>>>(
            Wcat + (size_t)i * WROW * NPK + (size_t)Dsz * NPK, proj, hbuf, cfc, ctr);
        // LIF gate + residual
        size_t nel = (size_t)NTOK * Dsz;
        lif_residual<<<(unsigned)((nel + 255) / 256), 256, 0, stream>>>(
            x, cfc, lif_thr + (size_t)i * Dsz, lif_lk + (size_t)i * Dsz,
            lif_st + (size_t)i * Dsz, nel);
        // ln2 -> f16
        layernorm_f16<<<NTOK / 8, 256, 0, stream>>>(x, ln2_s + (size_t)i * Dsz,
                                                    ln2_b + (size_t)i * Dsz, nrm);
        // mlp1 + exact GELU -> f16
        gemm_wmma_f16<EPI_GELU_F16, 0><<<dim3(Fsz / 128, NTOK / 64), 256, 0, stream>>>(
            nrm, W1h + (size_t)i * Dsz * Fsz, mlp_b1 + (size_t)i * Fsz,
            nullptr, nullptr, act1, NTOK, Fsz, Dsz);
        // mlp2 + residual add -> f32 (in-place on x)
        gemm_wmma_f16<EPI_RESID_F32, 0><<<dim3(Dsz / 128, NTOK / 64), 256, 0, stream>>>(
            act1, W2h + (size_t)i * Fsz * Dsz, mlp_b2 + (size_t)i * Dsz,
            x, x, nullptr, NTOK, Dsz, Fsz);
    }

    // final LN -> logits = nrm @ wte^T (wte stored (V,D) == B transposed)
    layernorm_f16<<<NTOK / 8, 256, 0, stream>>>(x, lnf_s, lnf_b, nrm);
    gemm_wmma_f16<EPI_F32, 1><<<dim3(Vsz / 128, NTOK / 64), 256, 0, stream>>>(
        nrm, wteh, nullptr, nullptr, out, nullptr, NTOK, Vsz, Dsz);

    // loss
    zero_u32<<<1, 32, 0, stream>>>((unsigned int*)lacc, 2);
    nll_loss<<<NTOK / 8, 256, 0, stream>>>(out, targets, lacc);
    loss_finalize<<<1, 1, 0, stream>>>(lacc, out + (size_t)NTOK * Vsz);
}
